// MedicalVQADecoder_81011673137306
// MI455X (gfx1250) — matrix operand order repaired and checked by reading, not compiled, not run on hardware
//
#include <hip/hip_runtime.h>
#include <hip/hip_bf16.h>

// ---------------------------------------------------------------------------
// Types for CDNA5 WMMA
// ---------------------------------------------------------------------------
typedef __bf16 bf16_t;
typedef __attribute__((ext_vector_type(16))) __bf16 v16bf;
typedef __attribute__((ext_vector_type(8)))  __bf16 v8bf;
typedef __attribute__((ext_vector_type(8)))  float  v8f;
typedef __attribute__((ext_vector_type(8)))  unsigned short v8u16;

#define TB 8
#define TS 512
#define TH 768
#define TNH 8
#define TDH 96
#define TFF 3072
#define TV 30000
#define TM (TB*TS)   // 4096 rows of activations

// ---------------------------------------------------------------------------
// Helpers
// ---------------------------------------------------------------------------
__device__ __forceinline__ unsigned short f2bf_u(float f) {
    unsigned u = __float_as_uint(f);
    unsigned r = u + 0x7fffu + ((u >> 16) & 1u);
    if ((u & 0x7fffffffu) > 0x7f800000u) r = u | 0x00400000u; // NaN
    return (unsigned short)(r >> 16);
}

__device__ __forceinline__ float gelu_f(float v) {
    return 0.5f * v * (1.0f + erff(v * 0.7071067811865475f));
}

// gfx1250 async memory->LDS copy (ASYNCcnt-tracked, no VGPR data movement).
// LDS byte address = low 32 bits of the generic pointer (ISA 10.2).
__device__ __forceinline__ void async_ld_b128(const void* gaddr, unsigned lds_addr) {
    asm volatile("global_load_async_to_lds_b128 %0, %1, off"
                 :: "v"(lds_addr), "v"(gaddr) : "memory");
}
__device__ __forceinline__ void wait_async_le3() {
    asm volatile("s_wait_asynccnt 3" ::: "memory");
}
__device__ __forceinline__ void wait_async_0() {
    asm volatile("s_wait_asynccnt 0" ::: "memory");
}
__device__ __forceinline__ unsigned lds_addr_of(const void* p) {
    return (unsigned)(size_t)p;
}

// ---------------------------------------------------------------------------
// Generic bf16 WMMA GEMM:  out[M,N] = A[M,K] @ Bw[N,K]^T (+bias)(gelu)(+resid)
// Block tile 64x128, K-step 32, 8 waves (2x4), each wave 32x32 via 2x2 WMMA.
// Double-buffered LDS staged with GLOBAL_LOAD_ASYNC_TO_LDS_B128; next tile's
// copies are in flight while the current tile feeds v_wmma_f32_16x16x32_bf16.
// ---------------------------------------------------------------------------
__global__ __launch_bounds__(256)
void gemm_bf16_kernel(const unsigned short* __restrict__ A,
                      const unsigned short* __restrict__ Bw,
                      float* __restrict__ outf,
                      unsigned short* __restrict__ outbf,
                      const float* __restrict__ bias,
                      const float* __restrict__ resid,
                      int M, int N, int K, int do_gelu)
{
    __shared__ unsigned short As[2][64 * 40]  __attribute__((aligned(16)));
    __shared__ unsigned short Bs[2][128 * 40] __attribute__((aligned(16)));

    const int tid   = threadIdx.x;
    const int lane  = tid & 31;
    const int wid   = tid >> 5;
    const int wm    = wid >> 2;   // 0..1
    const int wn    = wid & 3;    // 0..3
    const int bm0   = blockIdx.y * 64;
    const int bn0   = blockIdx.x * 128;
    const int laneM = lane & 15;
    const int hi    = lane >> 4;  // 0 or 1

    v8f acc[2][2];
    for (int i = 0; i < 2; i++)
        for (int j = 0; j < 2; j++)
            acc[i][j] = (v8f){0.f,0.f,0.f,0.f,0.f,0.f,0.f,0.f};

    // per-thread tile-staging coordinates
    const int ar = tid >> 2, ac = (tid & 3) * 8;   // A: 64 rows x 32 cols, 8 elem/thread
    const int br = tid >> 1, bc = (tid & 1) * 16;  // B: 128 rows x 32 cols, 16 elem/thread
    const int am = bm0 + ar;
    const int bn = bn0 + br;

    const int nk = K >> 5;

    // Issue 3 async b128 copies (1 A, 2 B) for K-tile kt into buffer buf.
    auto issue_tile = [&](int kt, int buf) {
        const int k0 = kt << 5;
        if (am < M) {
            async_ld_b128(A + (size_t)am * K + k0 + ac,
                          lds_addr_of(&As[buf][ar * 40 + ac]));
        } else {
            v8u16 z = {};
            *reinterpret_cast<v8u16*>(&As[buf][ar * 40 + ac]) = z;
        }
        if (bn < N) {
            const unsigned short* bp = Bw + (size_t)bn * K + k0 + bc;
            async_ld_b128(bp,     lds_addr_of(&Bs[buf][br * 40 + bc]));
            async_ld_b128(bp + 8, lds_addr_of(&Bs[buf][br * 40 + bc + 8]));
        } else {
            v8u16 z = {};
            *reinterpret_cast<v8u16*>(&Bs[buf][br * 40 + bc])     = z;
            *reinterpret_cast<v8u16*>(&Bs[buf][br * 40 + bc + 8]) = z;
        }
    };

    issue_tile(0, 0);

    for (int kt = 0; kt < nk; ++kt) {
        const int cur = kt & 1;
        if (kt + 1 < nk) {
            issue_tile(kt + 1, cur ^ 1);  // prefetch next tile into other buffer
            wait_async_le3();             // in-order completion => current tile resident
        } else {
            wait_async_0();
        }
        __syncthreads();  // all waves' copies (async + zero-fill ds stores) visible

        const unsigned short* Asb = As[cur];
        const unsigned short* Bsb = Bs[cur];

        // A fragment (16x32 bf16): lanes 0-15 row M, K in [0,8)+[16,24);
        // lanes 16-31 same rows, K in [8,16)+[24,32)
        v16bf af[2], bfr[2];
        #pragma unroll
        for (int i = 0; i < 2; i++) {
            const int r = wm * 32 + i * 16 + laneM;
            const v8bf lo = *reinterpret_cast<const v8bf*>(&Asb[r * 40 + hi * 8]);
            const v8bf hh = *reinterpret_cast<const v8bf*>(&Asb[r * 40 + hi * 8 + 16]);
            af[i] = __builtin_shufflevector(lo, hh, 0,1,2,3,4,5,6,7,8,9,10,11,12,13,14,15);
        }
        // B fragment (32x16 bf16): lane = column n, lanes 0-15 K in [0,8)+[16,24),
        // lanes 16-31 K in [8,16)+[24,32); column n of B == row n of Bw.
        #pragma unroll
        for (int j = 0; j < 2; j++) {
            const int r = wn * 32 + j * 16 + laneM;
            const v8bf lo = *reinterpret_cast<const v8bf*>(&Bsb[r * 40 + hi * 8]);
            const v8bf hh = *reinterpret_cast<const v8bf*>(&Bsb[r * 40 + hi * 8 + 16]);
            bfr[j] = __builtin_shufflevector(lo, hh, 0,1,2,3,4,5,6,7,8,9,10,11,12,13,14,15);
        }

        #pragma unroll
        for (int i = 0; i < 2; i++)
            #pragma unroll
            for (int j = 0; j < 2; j++)
                acc[i][j] = __builtin_amdgcn_wmma_f32_16x16x32_bf16(
                    false, af[i], false, bfr[j], (short)0, acc[i][j], false, false);

        __syncthreads();  // done reading before this buffer is overwritten
    }

    // Epilogue. C layout: VGPR r -> (M = r + 8*hi, N = laneM) within 16x16 tile.
    #pragma unroll
    for (int i = 0; i < 2; i++) {
        #pragma unroll
        for (int j = 0; j < 2; j++) {
            const int ng  = bn0 + wn * 32 + j * 16 + laneM;
            const int mg0 = bm0 + wm * 32 + i * 16 + hi * 8;
            if (ng < N) {
                const float bsv = bias ? bias[ng] : 0.f;
                #pragma unroll
                for (int r = 0; r < 8; r++) {
                    const int mg = mg0 + r;
                    if (mg < M) {
                        float v = acc[i][j][r] + bsv;
                        if (do_gelu) v = gelu_f(v);
                        const size_t idx = (size_t)mg * N + ng;
                        if (resid) v += resid[idx];
                        if (outf)  outf[idx]  = v;
                        if (outbf) outbf[idx] = f2bf_u(v);
                    }
                }
            }
        }
    }
}

// ---------------------------------------------------------------------------
// Causal self-attention. qkv: [M, 3H] rows = (b*S+s), [q | k | v] each H.
// One wave32 per query row; 8 waves per block. Output bf16 [M, H].
// ---------------------------------------------------------------------------
__global__ __launch_bounds__(256)
void attn_kernel(const float* __restrict__ qkv, unsigned short* __restrict__ outbf)
{
    __shared__ float sQ[8][TDH];
    __shared__ float sP[8][TS];

    const int tid  = threadIdx.x;
    const int lane = tid & 31;
    const int wid  = tid >> 5;

    const int idx = blockIdx.x * 8 + wid;   // ((b*NH + h)*S + q)
    const int q   = idx & (TS - 1);
    const int bh  = idx >> 9;
    const int h   = bh & (TNH - 1);
    const int b   = bh >> 3;

    const float* qrow = qkv + ((size_t)(b * TS + q)) * (3 * TH) + h * TDH;
    sQ[wid][lane]      = qrow[lane];
    sQ[wid][lane + 32] = qrow[lane + 32];
    sQ[wid][lane + 64] = qrow[lane + 64];
    __syncthreads();

    const float scale = 0.1020620726159658f;  // 1/sqrt(96)
    const float* kbase = qkv + (size_t)(b * TS) * (3 * TH) + TH + h * TDH;

    float sc[16];
    float mx = -3.0e38f;
    #pragma unroll 4
    for (int t = 0; t < 16; t++) {
        const int j = lane + 32 * t;
        float s = -1.0e30f;
        if (j <= q) {
            const float* kr = kbase + (size_t)j * (3 * TH);
            float d = 0.f;
            for (int c = 0; c < TDH; c++) d += sQ[wid][c] * kr[c];
            s = d * scale;
        }
        sc[t] = s;
        mx = fmaxf(mx, s);
    }
    for (int m = 16; m >= 1; m >>= 1) mx = fmaxf(mx, __shfl_xor(mx, m, 32));

    float sum = 0.f;
    #pragma unroll
    for (int t = 0; t < 16; t++) {
        const int j = lane + 32 * t;
        float p = __expf(sc[t] - mx);
        if (j > q) p = 0.f;
        sP[wid][j] = p;
        sum += p;
    }
    for (int m = 16; m >= 1; m >>= 1) sum += __shfl_xor(sum, m, 32);
    __syncthreads();

    const float inv = 1.0f / sum;
    const float* vbase = qkv + (size_t)(b * TS) * (3 * TH) + 2 * TH + h * TDH;
    float a0 = 0.f, a1 = 0.f, a2 = 0.f;
    for (int j = 0; j <= q; j++) {
        const float p = sP[wid][j];
        const float* vr = vbase + (size_t)j * (3 * TH);
        a0 += p * vr[lane];
        a1 += p * vr[lane + 32];
        a2 += p * vr[lane + 64];
    }
    unsigned short* orow = outbf + ((size_t)(b * TS + q)) * TH + h * TDH;
    orow[lane]      = f2bf_u(a0 * inv);
    orow[lane + 32] = f2bf_u(a1 * inv);
    orow[lane + 64] = f2bf_u(a2 * inv);
}

// ---------------------------------------------------------------------------
// LayerNorm over H=768, one row per block; bf16 output for the next GEMM.
// ---------------------------------------------------------------------------
__global__ __launch_bounds__(256)
void ln_bf16_kernel(const float* __restrict__ x, const float* __restrict__ g,
                    const float* __restrict__ bta, unsigned short* __restrict__ out)
{
    __shared__ float red[256];
    const int row = blockIdx.x;
    const int c = threadIdx.x;
    const float* xr = x + (size_t)row * TH;
    const float v0 = xr[c], v1 = xr[c + 256], v2 = xr[c + 512];

    red[c] = v0 + v1 + v2;
    __syncthreads();
    for (int st = 128; st > 0; st >>= 1) {
        if (c < st) red[c] += red[c + st];
        __syncthreads();
    }
    const float mean = red[0] * (1.0f / TH);
    __syncthreads();

    const float d0 = v0 - mean, d1 = v1 - mean, d2 = v2 - mean;
    red[c] = d0 * d0 + d1 * d1 + d2 * d2;
    __syncthreads();
    for (int st = 128; st > 0; st >>= 1) {
        if (c < st) red[c] += red[c + st];
        __syncthreads();
    }
    const float rstd = rsqrtf(red[0] * (1.0f / TH) + 1e-5f);

    unsigned short* orow = out + (size_t)row * TH;
    orow[c]       = f2bf_u(d0 * rstd * g[c]       + bta[c]);
    orow[c + 256] = f2bf_u(d1 * rstd * g[c + 256] + bta[c + 256]);
    orow[c + 512] = f2bf_u(d2 * rstd * g[c + 512] + bta[c + 512]);
}

// ---------------------------------------------------------------------------
// Embedding gather, f32->bf16 convert, broadcast add (cross-attn residual)
// ---------------------------------------------------------------------------
__global__ __launch_bounds__(256)
void embed_kernel(const int* __restrict__ tgt, const float* __restrict__ emb,
                  float* __restrict__ x)
{
    const int row = blockIdx.x;
    const int id = tgt[row];
    const float* er = emb + (size_t)id * TH;
    float* xr = x + (size_t)row * TH;
    for (int c = threadIdx.x; c < TH; c += 256) xr[c] = er[c];
}

__global__ __launch_bounds__(256)
void f2bf_kernel(const float* __restrict__ in, unsigned short* __restrict__ out, long n)
{
    for (long i = (long)blockIdx.x * blockDim.x + threadIdx.x; i < n;
         i += (long)gridDim.x * blockDim.x)
        out[i] = f2bf_u(in[i]);
}

__global__ __launch_bounds__(256)
void bcast_add_kernel(float* __restrict__ x, const float* __restrict__ y)
{
    const long n = (long)TM * TH;
    for (long i = (long)blockIdx.x * blockDim.x + threadIdx.x; i < n;
         i += (long)gridDim.x * blockDim.x) {
        const int b = (int)(i / ((long)TS * TH));
        const int c = (int)(i % TH);
        x[i] += y[b * TH + c];
    }
}

// out[b,n] = bias[n] + dot(xin[b,:], W[n,:])   (small: B=8 rows)
__global__ __launch_bounds__(256)
void vec_mat_kernel(const float* __restrict__ xin, const float* __restrict__ W,
                    const float* __restrict__ bias, float* __restrict__ out,
                    int N_, int K_)
{
    const int b = blockIdx.x;
    const float* xr = xin + (size_t)b * K_;
    for (int n = threadIdx.x; n < N_; n += 256) {
        const float* wr = W + (size_t)n * K_;
        float s = bias ? bias[n] : 0.f;
        for (int k = 0; k < K_; k++) s += xr[k] * wr[k];
        out[(size_t)b * N_ + n] = s;
    }
}

// Classifier head: logits_closed[b, 0..1]
__global__ __launch_bounds__(256)
void cls_kernel(const float* __restrict__ fused, const float* __restrict__ w1,
                const float* __restrict__ b1, const float* __restrict__ w2,
                const float* __restrict__ b2, float* __restrict__ out)
{
    __shared__ float sH[512];
    const int b = blockIdx.x;
    const float* xr = fused + (size_t)b * TH;
    for (int n = threadIdx.x; n < 512; n += 256) {
        const float* wr = w1 + (size_t)n * TH;
        float s = b1[n];
        for (int k = 0; k < TH; k++) s += xr[k] * wr[k];
        sH[n] = gelu_f(s);
    }
    __syncthreads();
    if (threadIdx.x < 2) {
        const float* wr = w2 + (size_t)threadIdx.x * 512;
        float s = b2[threadIdx.x];
        for (int k = 0; k < 512; k++) s += sH[k] * wr[k];
        out[b * 2 + threadIdx.x] = s;
    }
}

// ---------------------------------------------------------------------------
// Host orchestration
// ---------------------------------------------------------------------------
extern "C" void kernel_launch(void* const* d_in, const int* in_sizes, int n_in,
                              void* d_out, int out_size, void* d_ws, size_t ws_size,
                              hipStream_t stream)
{
    (void)in_sizes; (void)n_in; (void)out_size; (void)ws_size;

    const float* fused  = (const float*)d_in[0];
    const int*   tgt    = (const int*)  d_in[1];
    const float* emb    = (const float*)d_in[2];
    const float* cls_w1 = (const float*)d_in[3];
    const float* cls_b1 = (const float*)d_in[4];
    const float* cls_w2 = (const float*)d_in[5];
    const float* cls_b2 = (const float*)d_in[6];
    auto L = [&](int l, int j) -> const float* {
        return (const float*)d_in[7 + l * 18 + j];
    };
    // j: 0 ln1_g 1 ln1_b 2 ln2_g 3 ln2_b 4 ln3_g 5 ln3_b
    //    6 sa_w_in 7 sa_b_in 8 sa_w_out 9 sa_b_out
    //    10 ca_w_in 11 ca_b_in 12 ca_w_out 13 ca_b_out
    //    14 ff_w1 15 ff_b1 16 ff_w2 17 ff_b2

    // workspace layout
    char* wsp = (char*)d_ws;
    size_t off = 0;
    auto alloc = [&](size_t bytes) -> void* {
        void* p = wsp + off;
        off += (bytes + 255) & ~(size_t)255;
        return p;
    };
    float*          x      = (float*)         alloc((size_t)TM * TH * 4);
    float*          qkv    = (float*)         alloc((size_t)TM * 3 * TH * 4);
    unsigned short* abuf   = (unsigned short*)alloc((size_t)TM * TH * 2);
    unsigned short* ff1buf = (unsigned short*)alloc((size_t)TM * TFF * 2);
    unsigned short* wbuf   = (unsigned short*)alloc((size_t)TFF * TH * 2);
    unsigned short* embbf  = (unsigned short*)alloc((size_t)TV * TH * 2);
    float*          vc     = (float*)         alloc((size_t)TB * TH * 4);
    float*          yb     = (float*)         alloc((size_t)TB * TH * 4);

    float* out_closed = (float*)d_out;         // 8*2
    float* out_open   = (float*)d_out + 16;    // 4096*30000

    // classifier head (independent of decoder)
    cls_kernel<<<TB, 256, 0, stream>>>(fused, cls_w1, cls_b1, cls_w2, cls_b2, out_closed);

    // x = emb[target_ids]; emb -> bf16 for the final vocab GEMM
    embed_kernel<<<TM, 256, 0, stream>>>(tgt, emb, x);
    f2bf_kernel<<<4096, 256, 0, stream>>>(emb, embbf, (long)TV * TH);

    const dim3 blk(256);
    for (int l = 0; l < 3; l++) {
        // ---- self-attention ----
        ln_bf16_kernel<<<TM, blk, 0, stream>>>(x, L(l, 0), L(l, 1), abuf);
        f2bf_kernel<<<2048, blk, 0, stream>>>(L(l, 6), wbuf, (long)3 * TH * TH);
        gemm_bf16_kernel<<<dim3((3 * TH + 127) / 128, TM / 64), blk, 0, stream>>>(
            abuf, wbuf, qkv, nullptr, L(l, 7), nullptr, TM, 3 * TH, TH, 0);
        attn_kernel<<<TM * TNH / 8, blk, 0, stream>>>(qkv, abuf);
        f2bf_kernel<<<1024, blk, 0, stream>>>(L(l, 8), wbuf, (long)TH * TH);
        gemm_bf16_kernel<<<dim3(TH / 128, TM / 64), blk, 0, stream>>>(
            abuf, wbuf, x, nullptr, L(l, 9), x, TM, TH, TH, 0);

        // ---- cross-attention: Sk==1 -> softmax==1 -> out = Wout(v_c)+b broadcast ----
        vec_mat_kernel<<<TB, blk, 0, stream>>>(
            fused, L(l, 10) + (size_t)2 * TH * TH, L(l, 11) + 2 * TH, vc, TH, TH);
        vec_mat_kernel<<<TB, blk, 0, stream>>>(vc, L(l, 12), L(l, 13), yb, TH, TH);
        bcast_add_kernel<<<2048, blk, 0, stream>>>(x, yb);

        // ---- feed-forward ----
        ln_bf16_kernel<<<TM, blk, 0, stream>>>(x, L(l, 4), L(l, 5), abuf);
        f2bf_kernel<<<2048, blk, 0, stream>>>(L(l, 14), wbuf, (long)TFF * TH);
        gemm_bf16_kernel<<<dim3(TFF / 128, TM / 64), blk, 0, stream>>>(
            abuf, wbuf, nullptr, ff1buf, L(l, 15), nullptr, TM, TFF, TH, 1);
        f2bf_kernel<<<2048, blk, 0, stream>>>(L(l, 16), wbuf, (long)TH * TFF);
        gemm_bf16_kernel<<<dim3(TH / 128, TM / 64), blk, 0, stream>>>(
            ff1buf, wbuf, x, nullptr, L(l, 17), x, TM, TH, TFF, 0);
    }

    // ---- logits_open = x @ emb.T ----
    f2bf_kernel<<<1024, blk, 0, stream>>>(x, abuf, (long)TM * TH);
    gemm_bf16_kernel<<<dim3((TV + 127) / 128, TM / 64), blk, 0, stream>>>(
        abuf, embbf, out_open, nullptr, nullptr, nullptr, TM, TV, TH, 0);
}
